// Baller2Vec_49203145343285
// MI455X (gfx1250) — matrix operand order repaired and checked by reading, not compiled, not run on hardware
//
#include <hip/hip_runtime.h>

// ---------------------------------------------------------------------------
// Types for CDNA5 WMMA (gfx1250, wave32)
// ---------------------------------------------------------------------------
typedef __bf16 bf16;
typedef __attribute__((ext_vector_type(8)))  __bf16 bf16x8;
typedef __attribute__((ext_vector_type(16))) __bf16 v16bf;
typedef __attribute__((ext_vector_type(8)))  float  v8f;

#define P_ 10
#define T_ 20
#define S_ 240
#define B_ 128
#define D_ 512
#define H_ 8
#define HD_ 64
#define F_ 2048
#define L_ 6
#define C_ 9
#define M_TOK 30720          // B_*S_
#define SPAD 256             // padded S for V^T / P tiles
#define SCALE_F 22.62741699796952f   // sqrt(512)
#define HSCALE_F 0.125f              // 1/sqrt(64)

static __device__ __forceinline__ v8f zero8() {
  v8f z = {0.f,0.f,0.f,0.f,0.f,0.f,0.f,0.f};
  return z;
}

static __device__ __forceinline__ v8f wmma_bf16(v16bf a, v16bf b, v8f c) {
  return __builtin_amdgcn_wmma_f32_16x16x32_bf16(
      /*neg_a=*/false, a, /*neg_b=*/false, b,
      /*c_mod=*/(short)0, c, /*reuse_a=*/false, /*reuse_b=*/false);
}

// A fragment: 16x32 bf16 tile, row-major source with leading dim lda.
// ISA 16-bit A layout: lane m=lane&15 holds row m; half=lane>>4 selects
// K-halves {half*8 .. half*8+7} and {half*8+16 .. half*8+23} -> two 16B loads.
static __device__ __forceinline__ v16bf load_frag_a(const bf16* A, int lda,
                                                    int row0, int k0, int lane) {
  const int m = lane & 15, half = lane >> 4;
  const bf16* p = A + (size_t)(row0 + m) * lda + (k0 + half * 8);
  bf16x8 lo = *reinterpret_cast<const bf16x8*>(p);
  bf16x8 hi = *reinterpret_cast<const bf16x8*>(p + 16);
  return __builtin_shufflevector(lo, hi, 0,1,2,3,4,5,6,7,8,9,10,11,12,13,14,15);
}

// B fragment: 32x16 bf16 tile, but we load from B^T stored [N,K] row-major.
// ISA 16-bit B layout: lane n=lane&15 holds output column n, half=lane>>4
// selects K-halves {half*8 .. half*8+7} and {half*8+16 .. half*8+23} -> two 16B loads.
static __device__ __forceinline__ v16bf load_frag_b(const bf16* Bt, int ldb,
                                                    int col0, int k0, int lane) {
  const int n = lane & 15, half = lane >> 4;
  const bf16* p = Bt + (size_t)(col0 + n) * ldb + (k0 + half * 8);
  bf16x8 lo = *reinterpret_cast<const bf16x8*>(p);
  bf16x8 hi = *reinterpret_cast<const bf16x8*>(p + 16);
  return __builtin_shufflevector(lo, hi, 0,1,2,3,4,5,6,7,8,9,10,11,12,13,14,15);
}

// mask step: block-growing mask == causal ordering on "step" index
static __device__ __forceinline__ int mask_step(int i) {
  return i < P_ * T_ ? (i / P_) : (i < P_ * T_ + T_ ? i - P_ * T_ : i - P_ * T_ - T_);
}

// ---------------------------------------------------------------------------
// Generic WMMA GEMM:  C[M,N] = A[M,K] * Bt[N,K]^T  (+bias) + epilogue
// block 256 = 8 waves (2x4); block tile 128x256; wave tile 64x64 (4x4 C tiles)
// -> 16 b128 loads per 16 WMMAs per K-step (1.0 load/WMMA)
// MODE 0: QKV scatter (bias, write Q/K/Vt bf16)
// MODE 1: bias + in-place fp32 residual add into residF
// MODE 2: bias + ReLU -> bf16 outB
// MODE is a template parameter so each instantiation carries only its own
// epilogue (kills the runtime branch + dead pointers); launch_bounds(256,1)
// lifts the default VGPR cap so the 128 accumulator VGPRs never spill.
// ---------------------------------------------------------------------------
template <int MODE>
__global__ __launch_bounds__(256, 1) void gemm_bf16_epi(
    const bf16* __restrict__ A, int lda,
    const bf16* __restrict__ Bt, int ldb,
    const float* __restrict__ bias,
    int N, int Kdim,
    float* __restrict__ residF,
    bf16* __restrict__ outB,
    bf16* __restrict__ Qo, bf16* __restrict__ Ko, bf16* __restrict__ Vto) {
  const int tid  = threadIdx.x;
  const int lane = tid & 31;
  const int wave = tid >> 5;
  const int wm = wave >> 2;                     // 0..1
  const int wn = wave & 3;                      // 0..3
  const int rowBase = blockIdx.x * 128 + wm * 64;
  const int colBase = blockIdx.y * 256 + wn * 64;

  v8f acc[4][4];
#pragma unroll
  for (int i = 0; i < 4; ++i)
#pragma unroll
    for (int j = 0; j < 4; ++j) acc[i][j] = zero8();

  for (int k0 = 0; k0 < Kdim; k0 += 32) {
    v16bf af[4];
#pragma unroll
    for (int i = 0; i < 4; ++i)
      af[i] = load_frag_a(A, lda, rowBase + i * 16, k0, lane);
    v16bf bfg[4];
#pragma unroll
    for (int j = 0; j < 4; ++j)
      bfg[j] = load_frag_b(Bt, ldb, colBase + j * 16, k0, lane);
    if (k0 + 32 < Kdim) {  // gfx1250 global_prefetch_b8
      __builtin_prefetch(A + (size_t)(rowBase + (lane & 15)) * lda + k0 + 32, 0, 1);
      __builtin_prefetch(Bt + (size_t)(colBase + (lane & 15)) * ldb + k0 + 32, 0, 1);
    }
#pragma unroll
    for (int i = 0; i < 4; ++i)
#pragma unroll
      for (int j = 0; j < 4; ++j)
        acc[i][j] = wmma_bf16(af[i], bfg[j], acc[i][j]);
  }

  // epilogue; C layout: VGPR r of a tile = row (r + 8*half), lane&15 = col n
  const int n = lane & 15, half = lane >> 4;
#pragma unroll
  for (int i = 0; i < 4; ++i) {
#pragma unroll
    for (int j = 0; j < 4; ++j) {
      const int col = colBase + j * 16 + n;
      const float bval = bias[col];
      // mode-0 per-tile-uniform decode: col0 is 16-aligned within a
      // 64-aligned feature block, so part/h are tile-constant.
      const int part = col >> 9;         // 0=Q 1=K 2=V
      const int f = col & 511;
      const int h = f >> 6;
      const int d = f & 63;
#pragma unroll
      for (int r = 0; r < 8; ++r) {
        const int row = rowBase + i * 16 + r + half * 8;
        const float v = acc[i][j][r] + bval;
        if (MODE == 0) {
          const int b = row / S_;
          const int s = row - b * S_;
          if (part == 0)
            Qo[(((size_t)(b * H_ + h)) * S_ + s) * HD_ + d] = (bf16)v;
          else if (part == 1)
            Ko[(((size_t)(b * H_ + h)) * S_ + s) * HD_ + d] = (bf16)v;
          else
            Vto[(((size_t)(b * H_ + h)) * HD_ + d) * SPAD + s] = (bf16)v;
        } else if (MODE == 1) {
          const size_t idx = (size_t)row * N + col;
          residF[idx] = residF[idx] + v;
        } else {
          const size_t idx = (size_t)row * N + col;
          outB[idx] = (bf16)(v > 0.f ? v : 0.f);
        }
      }
    }
  }
}

// ---------------------------------------------------------------------------
// Attention: one block (4 waves) per (b,h). Each wave: 16-row Q tile,
// full 240-col scores in registers, register softmax (wave32 half-shuffles),
// P -> LDS -> WMMA A fragments, P*V with V stored transposed [HD, SPAD].
// ---------------------------------------------------------------------------
__global__ __launch_bounds__(128, 1) void attention_k(
    const bf16* __restrict__ Q, const bf16* __restrict__ K,
    const bf16* __restrict__ Vt, bf16* __restrict__ O) {
  __shared__ __align__(16) bf16 Pl[4][16][SPAD];
  const int tid  = threadIdx.x;
  const int lane = tid & 31;
  const int wave = tid >> 5;
  const int n = lane & 15, half = lane >> 4;
  const int bh = blockIdx.x;
  const int b = bh >> 3;
  const int h = bh & 7;
  const bf16* Qb = Q  + (size_t)bh * S_ * HD_;
  const bf16* Kb = K  + (size_t)bh * S_ * HD_;
  const bf16* Vb = Vt + (size_t)bh * HD_ * SPAD;

  for (int qt = wave; qt < 15; qt += 4) {
    const v16bf qf0 = load_frag_a(Qb, HD_, qt * 16, 0, lane);
    const v16bf qf1 = load_frag_a(Qb, HD_, qt * 16, 32, lane);

    v8f sc[15];
#pragma unroll
    for (int kt = 0; kt < 15; ++kt) {
      v8f a = zero8();
      a = wmma_bf16(qf0, load_frag_b(Kb, HD_, kt * 16, 0, lane), a);
      a = wmma_bf16(qf1, load_frag_b(Kb, HD_, kt * 16, 32, lane), a);
      sc[kt] = a;
    }

    // scale + additive mask (causal in step space)
#pragma unroll
    for (int kt = 0; kt < 15; ++kt) {
      const int cs = mask_step(kt * 16 + n);
#pragma unroll
      for (int r = 0; r < 8; ++r) {
        const int row = qt * 16 + r + half * 8;
        float v = sc[kt][r] * HSCALE_F;
        if (cs > mask_step(row)) v = -1e30f;
        sc[kt][r] = v;
      }
    }

    // softmax: rows live across the 16 lanes of each half-wave
    float rinv[8];
#pragma unroll
    for (int r = 0; r < 8; ++r) {
      float m = -3.0e38f;
#pragma unroll
      for (int kt = 0; kt < 15; ++kt) m = fmaxf(m, sc[kt][r]);
#pragma unroll
      for (int off = 1; off < 16; off <<= 1)
        m = fmaxf(m, __shfl_xor(m, off, 32));
      float ssum = 0.f;
#pragma unroll
      for (int kt = 0; kt < 15; ++kt) {
        const float e = __expf(sc[kt][r] - m);
        sc[kt][r] = e;
        ssum += e;
      }
#pragma unroll
      for (int off = 1; off < 16; off <<= 1)
        ssum += __shfl_xor(ssum, off, 32);
      rinv[r] = 1.f / ssum;
    }

    // write normalized P into this wave's private LDS tile (+zero pad cols)
#pragma unroll
    for (int r = 0; r < 8; ++r) {
      const int m = r + half * 8;
#pragma unroll
      for (int kt = 0; kt < 15; ++kt)
        Pl[wave][m][kt * 16 + n] = (bf16)(sc[kt][r] * rinv[r]);
      Pl[wave][m][S_ + n] = (bf16)0.f;
    }

    // P[16 x 256] @ V[256 x 64]   (pad region contributes zero)
    v8f oacc[4];
#pragma unroll
    for (int dt = 0; dt < 4; ++dt) oacc[dt] = zero8();
#pragma unroll
    for (int ks = 0; ks < 8; ++ks) {
      const bf16* pp = &Pl[wave][n][ks * 32 + half * 8];  // A frag: row=lane&15
      bf16x8 lo = *reinterpret_cast<const bf16x8*>(pp);
      bf16x8 hi = *reinterpret_cast<const bf16x8*>(pp + 16);
      v16bf pa = __builtin_shufflevector(lo, hi, 0,1,2,3,4,5,6,7,8,9,10,11,12,13,14,15);
#pragma unroll
      for (int dt = 0; dt < 4; ++dt)
        oacc[dt] = wmma_bf16(pa, load_frag_b(Vb, SPAD, dt * 16, ks * 32, lane), oacc[dt]);
    }

    // store O as [token, D] bf16 (ready as A-operand of the Wo GEMM)
#pragma unroll
    for (int dt = 0; dt < 4; ++dt)
#pragma unroll
      for (int r = 0; r < 8; ++r) {
        const int srow = qt * 16 + r + half * 8;
        O[((size_t)(b * S_ + srow)) * D_ + h * HD_ + dt * 16 + n] = (bf16)oacc[dt][r];
      }
  }
}

// ---------------------------------------------------------------------------
// LayerNorm (in-place on fp32 residual stream, also refresh bf16 copy)
// one 128-thread block per token row (D=512 -> 4 elems/thread)
// ---------------------------------------------------------------------------
__global__ __launch_bounds__(128) void layernorm_k(
    float* __restrict__ X, bf16* __restrict__ Xb,
    const float* __restrict__ g, const float* __restrict__ bta) {
  __shared__ float red[128];
  const int row = blockIdx.x;
  const int tid = threadIdx.x;
  float* xr = X + (size_t)row * D_;
  float v[4];
  float s = 0.f;
#pragma unroll
  for (int i = 0; i < 4; ++i) { v[i] = xr[tid + i * 128]; s += v[i]; }
  red[tid] = s;
  __syncthreads();
  for (int off = 64; off > 0; off >>= 1) {
    if (tid < off) red[tid] += red[tid + off];
    __syncthreads();
  }
  const float mean = red[0] * (1.f / (float)D_);
  __syncthreads();
  float vs = 0.f;
#pragma unroll
  for (int i = 0; i < 4; ++i) { const float d = v[i] - mean; vs += d * d; }
  red[tid] = vs;
  __syncthreads();
  for (int off = 64; off > 0; off >>= 1) {
    if (tid < off) red[tid] += red[tid + off];
    __syncthreads();
  }
  const float rstd = rsqrtf(red[0] * (1.f / (float)D_) + 1e-5f);
#pragma unroll
  for (int i = 0; i < 4; ++i) {
    const int c = tid + i * 128;
    const float o = (v[i] - mean) * rstd * g[c] + bta[c];
    xr[c] = o;
    Xb[(size_t)row * D_ + c] = (bf16)o;
  }
}

// ---------------------------------------------------------------------------
// Token construction: player/ball 2-layer MLPs (fp32) + cls copy.
// one 128-thread block per token.
// ---------------------------------------------------------------------------
__global__ __launch_bounds__(128) void build_tokens_k(
    const int* __restrict__ pidx, const float* __restrict__ pxs,
    const float* __restrict__ pys, const float* __restrict__ bxs,
    const float* __restrict__ bys, const float* __restrict__ emb,
    const float* __restrict__ ball_e, const float* __restrict__ cls_e,
    const float* __restrict__ pW1, const float* __restrict__ pb1,
    const float* __restrict__ pW2, const float* __restrict__ pb2,
    const float* __restrict__ bW1, const float* __restrict__ bb1,
    const float* __restrict__ bW2, const float* __restrict__ bb2,
    float* __restrict__ Xf, bf16* __restrict__ Xb) {
  __shared__ float in_s[66];
  __shared__ float h_s[128];
  const int token = blockIdx.x;
  const int tid = threadIdx.x;
  const int b = token / S_;
  const int s = token - b * S_;
  const int t = s / (P_ + 2);
  const int j = s - t * (P_ + 2);
  float* xr = Xf + (size_t)token * D_;
  bf16* xbr = Xb + (size_t)token * D_;

  if (j == P_ + 1) {  // cls token: raw embedding, no scale
#pragma unroll
    for (int i = 0; i < 4; ++i) {
      const int c = tid + i * 128;
      const float v = cls_e[c];
      xr[c] = v;
      xbr[c] = (bf16)v;
    }
    return;
  }
  const bool isPlayer = (j < P_);
  if (tid < 66) {
    float val;
    if (tid < 64) {
      if (isPlayer) {
        const int idx = pidx[(b * T_ + t) * P_ + j];
        val = emb[(size_t)idx * 64 + tid];
      } else {
        val = ball_e[tid];
      }
    } else if (tid == 64) {
      val = isPlayer ? pxs[(b * T_ + t) * P_ + j] : bxs[b * T_ + t];
    } else {
      val = isPlayer ? pys[(b * T_ + t) * P_ + j] : bys[b * T_ + t];
    }
    in_s[tid] = val;
  }
  __syncthreads();
  const float* W1 = isPlayer ? pW1 : bW1;
  const float* b1 = isPlayer ? pb1 : bb1;
  const float* W2 = isPlayer ? pW2 : bW2;
  const float* b2 = isPlayer ? pb2 : bb2;
  float hsum = b1[tid];
  for (int i = 0; i < 66; ++i) hsum += in_s[i] * W1[i * 128 + tid];
  h_s[tid] = hsum > 0.f ? hsum : 0.f;
  __syncthreads();
#pragma unroll
  for (int q = 0; q < 4; ++q) {
    const int c = tid + q * 128;
    float o = b2[c];
    for (int i = 0; i < 128; ++i) o += h_s[i] * W2[i * D_ + c];
    o *= SCALE_F;
    xr[c] = o;
    xbr[c] = (bf16)o;
  }
}

// fp32 [K,N] -> bf16 [N,K]  (weight transpose+convert, done once per launch)
__global__ void transpose_bf16_k(const float* __restrict__ src,
                                 bf16* __restrict__ dst, int K, int N) {
  const size_t idx = (size_t)blockIdx.x * 256 + threadIdx.x;
  if (idx >= (size_t)K * N) return;
  const int nn = (int)(idx / K);
  const int kk = (int)(idx - (size_t)nn * K);
  dst[idx] = (bf16)src[(size_t)kk * N + nn];
}

__global__ void zero_bf16_k(bf16* __restrict__ p, size_t count) {
  const size_t idx = (size_t)blockIdx.x * 256 + threadIdx.x;
  if (idx < count) p[idx] = (bf16)0.f;
}

// final classifier: logits of token S-1 only -> [B, C]
__global__ void classifier_k(const float* __restrict__ Xf,
                             const float* __restrict__ W,
                             const float* __restrict__ bias,
                             float* __restrict__ out) {
  const int idx = blockIdx.x * 256 + threadIdx.x;
  if (idx >= B_ * C_) return;
  const int b = idx / C_;
  const int c = idx - b * C_;
  const float* xr = Xf + ((size_t)(b * S_ + (S_ - 1))) * D_;
  float s = bias[c];
  for (int d = 0; d < D_; ++d) s += xr[d] * W[d * C_ + c];
  out[idx] = s;
}

// ---------------------------------------------------------------------------
extern "C" void kernel_launch(void* const* d_in, const int* in_sizes, int n_in,
                              void* d_out, int out_size, void* d_ws, size_t ws_size,
                              hipStream_t stream) {
  (void)in_sizes; (void)n_in; (void)out_size; (void)ws_size;
  const int*   pidx  = (const int*)  d_in[0];
  const float* pxs   = (const float*)d_in[1];
  const float* pys   = (const float*)d_in[2];
  const float* bxs   = (const float*)d_in[3];
  const float* bys   = (const float*)d_in[4];
  const float* emb   = (const float*)d_in[5];
  const float* balle = (const float*)d_in[6];
  const float* clse  = (const float*)d_in[7];
  const float* pW1   = (const float*)d_in[8];
  const float* pb1   = (const float*)d_in[9];
  const float* pW2   = (const float*)d_in[10];
  const float* pb2   = (const float*)d_in[11];
  const float* bW1   = (const float*)d_in[12];
  const float* bb1   = (const float*)d_in[13];
  const float* bW2   = (const float*)d_in[14];
  const float* bb2   = (const float*)d_in[15];
  const float* Wqkv  = (const float*)d_in[16];
  const float* bqkv  = (const float*)d_in[17];
  const float* Wo    = (const float*)d_in[18];
  const float* bo    = (const float*)d_in[19];
  const float* ln1g  = (const float*)d_in[20];
  const float* ln1b  = (const float*)d_in[21];
  const float* Wf1   = (const float*)d_in[22];
  const float* bf1   = (const float*)d_in[23];
  const float* Wf2   = (const float*)d_in[24];
  const float* bf2   = (const float*)d_in[25];
  const float* ln2g  = (const float*)d_in[26];
  const float* ln2b  = (const float*)d_in[27];
  const float* clsW  = (const float*)d_in[28];
  const float* clsb  = (const float*)d_in[29];

  char* ws = (char*)d_ws;
  size_t off = 0;
  auto alloc = [&](size_t bytes) -> void* {
    void* p = (void*)(ws + off);
    off += bytes;
    off = (off + 255) & ~(size_t)255;
    return p;
  };

  float* Xf   = (float*)alloc((size_t)M_TOK * D_ * 4);
  bf16*  Xb   = (bf16*) alloc((size_t)M_TOK * D_ * 2);
  bf16*  Qb   = (bf16*) alloc((size_t)B_ * H_ * S_ * HD_ * 2);
  bf16*  Kb   = (bf16*) alloc((size_t)B_ * H_ * S_ * HD_ * 2);
  bf16*  Vt   = (bf16*) alloc((size_t)B_ * H_ * HD_ * SPAD * 2);
  bf16*  Ob   = (bf16*) alloc((size_t)M_TOK * D_ * 2);
  bf16*  Hb   = (bf16*) alloc((size_t)M_TOK * F_ * 2);
  bf16*  WqkvT= (bf16*) alloc((size_t)L_ * 3 * D_ * D_ * 2);
  bf16*  WoT  = (bf16*) alloc((size_t)L_ * D_ * D_ * 2);
  bf16*  Wf1T = (bf16*) alloc((size_t)L_ * F_ * D_ * 2);
  bf16*  Wf2T = (bf16*) alloc((size_t)L_ * D_ * F_ * 2);

  // zero V^T so its padded columns (s=240..255) stay zero for every layer
  {
    const size_t cnt = (size_t)B_ * H_ * HD_ * SPAD;
    zero_bf16_k<<<(unsigned)((cnt + 255) / 256), 256, 0, stream>>>(Vt, cnt);
  }

  // token construction (fp32 MLPs + bf16 copy)
  build_tokens_k<<<M_TOK, 128, 0, stream>>>(pidx, pxs, pys, bxs, bys, emb, balle,
                                            clse, pW1, pb1, pW2, pb2, bW1, bb1,
                                            bW2, bb2, Xf, Xb);

  // weight transpose+convert to bf16 [N,K]
  for (int l = 0; l < L_; ++l) {
    transpose_bf16_k<<<(3 * D_ * D_ + 255) / 256, 256, 0, stream>>>(
        Wqkv + (size_t)l * D_ * 3 * D_, WqkvT + (size_t)l * 3 * D_ * D_, D_, 3 * D_);
    transpose_bf16_k<<<(D_ * D_ + 255) / 256, 256, 0, stream>>>(
        Wo + (size_t)l * D_ * D_, WoT + (size_t)l * D_ * D_, D_, D_);
    transpose_bf16_k<<<(D_ * F_ + 255) / 256, 256, 0, stream>>>(
        Wf1 + (size_t)l * D_ * F_, Wf1T + (size_t)l * F_ * D_, D_, F_);
    transpose_bf16_k<<<(F_ * D_ + 255) / 256, 256, 0, stream>>>(
        Wf2 + (size_t)l * F_ * D_, Wf2T + (size_t)l * D_ * F_, F_, D_);
  }

  const dim3 blk(256);
  for (int l = 0; l < L_; ++l) {
    // QKV projection -> scatter bf16 Q/K/V^T    (N=1536: grid y = 6)
    gemm_bf16_epi<0><<<dim3(M_TOK / 128, (3 * D_) / 256), blk, 0, stream>>>(
        Xb, D_, WqkvT + (size_t)l * 3 * D_ * D_, D_, bqkv + (size_t)l * 3 * D_,
        3 * D_, D_, nullptr, nullptr, Qb, Kb, Vt);

    // masked softmax attention
    attention_k<<<B_ * H_, 128, 0, stream>>>(Qb, Kb, Vt, Ob);

    // output projection + residual (in-place fp32)   (N=512: grid y = 2)
    gemm_bf16_epi<1><<<dim3(M_TOK / 128, D_ / 256), blk, 0, stream>>>(
        Ob, D_, WoT + (size_t)l * D_ * D_, D_, bo + (size_t)l * D_,
        D_, D_, Xf, nullptr, nullptr, nullptr, nullptr);

    layernorm_k<<<M_TOK, 128, 0, stream>>>(Xf, Xb, ln1g + (size_t)l * D_,
                                           ln1b + (size_t)l * D_);

    // FFN1 + ReLU -> bf16 hidden                 (N=2048: grid y = 8)
    gemm_bf16_epi<2><<<dim3(M_TOK / 128, F_ / 256), blk, 0, stream>>>(
        Xb, D_, Wf1T + (size_t)l * F_ * D_, D_, bf1 + (size_t)l * F_,
        F_, D_, nullptr, Hb, nullptr, nullptr, nullptr);

    // FFN2 + residual (in-place fp32)            (N=512: grid y = 2)
    gemm_bf16_epi<1><<<dim3(M_TOK / 128, D_ / 256), blk, 0, stream>>>(
        Hb, F_, Wf2T + (size_t)l * D_ * F_, F_, bf2 + (size_t)l * D_,
        D_, F_, Xf, nullptr, nullptr, nullptr, nullptr);

    layernorm_k<<<M_TOK, 128, 0, stream>>>(Xf, Xb, ln2g + (size_t)l * D_,
                                           ln2b + (size_t)l * D_);
  }

  classifier_k<<<(B_ * C_ + 255) / 256, 256, 0, stream>>>(Xf, clsW, clsb,
                                                          (float*)d_out);
}
